// GroupedExperts_63084479643707
// MI455X (gfx1250) — compile-verified, matrix operand order checked by reading, not run on hardware
//
#include <hip/hip_runtime.h>
#include <hip/hip_bf16.h>

#define E_ 8
#define D_ 1024
#define H_ 2816
#define T_ 16384
#define BM 128
#define BN 128
#define BK 32
#define LDT 40            // LDS row stride in bf16 elements (32 + 8 pad -> 80B = 20 banks)
#define TILE_ELEMS (BM * LDT)
#define MAX_TILES 136     // sum(ceil(c_e/128)) <= (T-64)/128 + 8 = 135

typedef __bf16 bf16;
typedef __attribute__((ext_vector_type(4)))  bf16  bf16x4;
typedef __attribute__((ext_vector_type(8)))  bf16  bf16x8;
typedef __attribute__((ext_vector_type(16))) bf16  bf16x16;
typedef __attribute__((ext_vector_type(8)))  float f32x8;

// ---------------------------------------------------------------------------
// Tile table: one entry per 128-row tile of a single expert's token range.
// tinfo[0] = n_tiles ; tinfo[16 + 4*t + {0,1,2}] = {expert, row_start, rows}
// ---------------------------------------------------------------------------
__global__ void build_tiles_k(const int* __restrict__ counts, int* __restrict__ tinfo) {
    if (threadIdx.x != 0 || blockIdx.x != 0) return;
    int off = 0, nt = 0;
    for (int e = 0; e < E_; ++e) {
        int c = counts[e];
        for (int i = 0; i < c; i += BM) {
            tinfo[16 + nt * 4 + 0] = e;
            tinfo[16 + nt * 4 + 1] = off + i;
            int rem = c - i;
            tinfo[16 + nt * 4 + 2] = rem < BM ? rem : BM;
            ++nt;
        }
        off += c;
    }
    tinfo[0] = nt;
}

__global__ void zero_out_k(float4* __restrict__ o) {
    size_t i = (size_t)blockIdx.x * blockDim.x + threadIdx.x;
    o[i] = make_float4(0.f, 0.f, 0.f, 0.f);
}

// ---------------------------------------------------------------------------
// Register-staged pipeline: load() issues global loads into VGPRs (no wait in
// the WMMA shadow), commit() converts + stores to LDS a full K-step later.
// Rows >= valid_rows are zero-filled so WMMA needs no masking.
// ---------------------------------------------------------------------------
struct RegsF32  { float4 v[4]; };   // 128x32 fp32 tile slice per thread
struct RegsBF16 { bf16x8 v[2]; };   // 128x32 bf16 tile slice per thread

__device__ __forceinline__ RegsF32 load_f32(const float* __restrict__ base, int ld,
                                            int valid_rows, int tid) {
    RegsF32 r;
#pragma unroll
    for (int j = 0; j < 4; ++j) {           // 1024 float4 slots / 256 thr
        int q   = tid + j * 256;
        int row = q >> 3;                   // 8 float4 per row
        int col = (q & 7) << 2;
        float4 v = make_float4(0.f, 0.f, 0.f, 0.f);
        if (row < valid_rows) v = *(const float4*)(base + (size_t)row * ld + col);
        r.v[j] = v;
    }
    return r;
}

__device__ __forceinline__ void commit_f32(const RegsF32& r, bf16* __restrict__ s, int tid) {
#pragma unroll
    for (int j = 0; j < 4; ++j) {
        int q   = tid + j * 256;
        int row = q >> 3;
        int col = (q & 7) << 2;
        bf16x4 p;
        p[0] = (bf16)r.v[j].x; p[1] = (bf16)r.v[j].y;
        p[2] = (bf16)r.v[j].z; p[3] = (bf16)r.v[j].w;
        *(bf16x4*)(&s[row * LDT + col]) = p;
    }
}

__device__ __forceinline__ RegsBF16 load_bf16(const bf16* __restrict__ base, int ld,
                                              int valid_rows, int tid) {
    RegsBF16 r;
#pragma unroll
    for (int j = 0; j < 2; ++j) {           // 512 bf16x8 slots / 256 thr
        int q   = tid + j * 256;
        int row = q >> 2;                   // 4 bf16x8 per row
        int col = (q & 3) << 3;
        bf16x8 v = {};
        if (row < valid_rows) v = *(const bf16x8*)(base + (size_t)row * ld + col);
        r.v[j] = v;
    }
    return r;
}

__device__ __forceinline__ void commit_bf16(const RegsBF16& r, bf16* __restrict__ s, int tid) {
#pragma unroll
    for (int j = 0; j < 2; ++j) {
        int q   = tid + j * 256;
        int row = q >> 2;
        int col = (q & 3) << 3;
        *(bf16x8*)(&s[row * LDT + col]) = r.v[j];
    }
}

// A/B fragment for v_wmma_f32_16x16x32_bf16 (symmetric TN layout):
// lane l (0..15): row = l, K 0..7 & 16..23 ; lane 16..31: same row, K+8.
__device__ __forceinline__ bf16x16 load_frag(const bf16* __restrict__ s, int row, int khalf) {
    bf16x8 lo = *(const bf16x8*)(s + row * LDT + khalf);
    bf16x8 hi = *(const bf16x8*)(s + row * LDT + khalf + 16);
    return __builtin_shufflevector(lo, hi, 0, 1, 2, 3, 4, 5, 6, 7,
                                           8, 9, 10, 11, 12, 13, 14, 15);
}

#define WMMA_BF16(A, B, C) \
    __builtin_amdgcn_wmma_f32_16x16x32_bf16(false, (A), false, (B), (short)0, (C), false, false)

// ---------------------------------------------------------------------------
// Phase 1: h = silu(x @ w1[e]^T) * (x @ w3[e]^T)   -> h[T,H] bf16
// grid = (MAX_TILES, H/128), block = 256 (8 waves, 2x4 wave grid)
// Double-buffered LDS + register-staged global prefetch.
// ---------------------------------------------------------------------------
__global__ __launch_bounds__(256) void moe_gemm13(const float* __restrict__ x,
                                                  const float* __restrict__ w1,
                                                  const float* __restrict__ w3,
                                                  const int* __restrict__ tinfo,
                                                  bf16* __restrict__ h) {
    const int nt = tinfo[0];
    if ((int)blockIdx.x >= nt) return;
    const int e    = tinfo[16 + blockIdx.x * 4 + 0];
    const int rs   = tinfo[16 + blockIdx.x * 4 + 1];
    const int rows = tinfo[16 + blockIdx.x * 4 + 2];
    const int nb   = blockIdx.y * BN;

    __shared__ bf16 sA[2][TILE_ELEMS];
    __shared__ bf16 sB1[2][TILE_ELEMS];
    __shared__ bf16 sB3[2][TILE_ELEMS];

    const float* xa = x  + (size_t)rs * D_;
    const float* b1 = w1 + (size_t)e * H_ * D_ + (size_t)nb * D_;
    const float* b3 = w3 + (size_t)e * H_ * D_ + (size_t)nb * D_;

    const int tid  = threadIdx.x;
    const int wave = tid >> 5, lane = tid & 31;
    const int wm = wave >> 2, wn = wave & 3;        // 2 x 4 wave grid
    const int lrow  = lane & 15;
    const int khalf = (lane >> 4) << 3;             // 0 or 8
    const int mofs  = (lane >> 4) << 3;             // C layout M offset

    f32x8 acc1[4][2], acc3[4][2];
#pragma unroll
    for (int i = 0; i < 4; ++i)
#pragma unroll
        for (int j = 0; j < 2; ++j) { acc1[i][j] = (f32x8){}; acc3[i][j] = (f32x8){}; }

    // pipeline prologue: tile 0 -> LDS buf0, tile 1 -> registers
    RegsF32 rA  = load_f32(xa, D_, rows, tid);
    RegsF32 rB1 = load_f32(b1, D_, BM,   tid);
    RegsF32 rB3 = load_f32(b3, D_, BM,   tid);
    commit_f32(rA,  sA[0],  tid);
    commit_f32(rB1, sB1[0], tid);
    commit_f32(rB3, sB3[0], tid);
    rA  = load_f32(xa + BK, D_, rows, tid);
    rB1 = load_f32(b1 + BK, D_, BM,   tid);
    rB3 = load_f32(b3 + BK, D_, BM,   tid);
    __syncthreads();

    int buf = 0;
    for (int k0 = 0; k0 < D_; k0 += BK, buf ^= 1) {
        bf16x16 af[4], bfr1[2], bfr3[2];
#pragma unroll
        for (int ms = 0; ms < 4; ++ms)
            af[ms] = load_frag(sA[buf], wm * 64 + ms * 16 + lrow, khalf);
#pragma unroll
        for (int ns = 0; ns < 2; ++ns) {
            bfr1[ns] = load_frag(sB1[buf], wn * 32 + ns * 16 + lrow, khalf);
            bfr3[ns] = load_frag(sB3[buf], wn * 32 + ns * 16 + lrow, khalf);
        }
#pragma unroll
        for (int ms = 0; ms < 4; ++ms)
#pragma unroll
            for (int ns = 0; ns < 2; ++ns) {
                acc1[ms][ns] = WMMA_BF16(af[ms], bfr1[ns], acc1[ms][ns]);
                acc3[ms][ns] = WMMA_BF16(af[ms], bfr3[ns], acc3[ms][ns]);
            }

        // commit prefetched tile (loadcnt wait lands after the WMMAs above),
        // then issue global loads for the tile after that
        if (k0 + BK < D_) {
            commit_f32(rA,  sA[buf ^ 1],  tid);
            commit_f32(rB1, sB1[buf ^ 1], tid);
            commit_f32(rB3, sB3[buf ^ 1], tid);
            if (k0 + 2 * BK < D_) {
                rA  = load_f32(xa + k0 + 2 * BK, D_, rows, tid);
                rB1 = load_f32(b1 + k0 + 2 * BK, D_, BM,   tid);
                rB3 = load_f32(b3 + k0 + 2 * BK, D_, BM,   tid);
            }
        }
        __syncthreads();
    }

    // Fused SwiGLU epilogue -> bf16 h
#pragma unroll
    for (int ms = 0; ms < 4; ++ms)
#pragma unroll
        for (int ns = 0; ns < 2; ++ns)
#pragma unroll
            for (int r = 0; r < 8; ++r) {
                int ml = wm * 64 + ms * 16 + mofs + r;
                if (ml < rows) {
                    float v1 = acc1[ms][ns][r];
                    float v3 = acc3[ms][ns][r];
                    float hv = (v1 / (1.0f + __expf(-v1))) * v3;
                    h[(size_t)(rs + ml) * H_ + (nb + wn * 32 + ns * 16 + lrow)] = (bf16)hv;
                }
            }
}

// ---------------------------------------------------------------------------
// Phase 2: out = h @ w2[e]^T   -> out[T,D] fp32
// grid = (MAX_TILES, D/128), block = 256, same pipeline
// ---------------------------------------------------------------------------
__global__ __launch_bounds__(256) void moe_gemm2(const bf16* __restrict__ h,
                                                 const float* __restrict__ w2,
                                                 const int* __restrict__ tinfo,
                                                 float* __restrict__ out) {
    const int nt = tinfo[0];
    if ((int)blockIdx.x >= nt) return;
    const int e    = tinfo[16 + blockIdx.x * 4 + 0];
    const int rs   = tinfo[16 + blockIdx.x * 4 + 1];
    const int rows = tinfo[16 + blockIdx.x * 4 + 2];
    const int nb   = blockIdx.y * BN;

    __shared__ bf16 sA[2][TILE_ELEMS];
    __shared__ bf16 sB[2][TILE_ELEMS];

    const bf16*  ha = h  + (size_t)rs * H_;
    const float* bw = w2 + (size_t)e * D_ * H_ + (size_t)nb * H_;

    const int tid  = threadIdx.x;
    const int wave = tid >> 5, lane = tid & 31;
    const int wm = wave >> 2, wn = wave & 3;
    const int lrow  = lane & 15;
    const int khalf = (lane >> 4) << 3;
    const int mofs  = (lane >> 4) << 3;

    f32x8 acc[4][2];
#pragma unroll
    for (int i = 0; i < 4; ++i)
#pragma unroll
        for (int j = 0; j < 2; ++j) acc[i][j] = (f32x8){};

    RegsBF16 rA = load_bf16(ha, H_, rows, tid);
    RegsF32  rB = load_f32 (bw, H_, BM,   tid);
    commit_bf16(rA, sA[0], tid);
    commit_f32 (rB, sB[0], tid);
    rA = load_bf16(ha + BK, H_, rows, tid);
    rB = load_f32 (bw + BK, H_, BM,   tid);
    __syncthreads();

    int buf = 0;
    for (int k0 = 0; k0 < H_; k0 += BK, buf ^= 1) {
        bf16x16 af[4], bfr[2];
#pragma unroll
        for (int ms = 0; ms < 4; ++ms)
            af[ms] = load_frag(sA[buf], wm * 64 + ms * 16 + lrow, khalf);
#pragma unroll
        for (int ns = 0; ns < 2; ++ns)
            bfr[ns] = load_frag(sB[buf], wn * 32 + ns * 16 + lrow, khalf);
#pragma unroll
        for (int ms = 0; ms < 4; ++ms)
#pragma unroll
            for (int ns = 0; ns < 2; ++ns)
                acc[ms][ns] = WMMA_BF16(af[ms], bfr[ns], acc[ms][ns]);

        if (k0 + BK < H_) {
            commit_bf16(rA, sA[buf ^ 1], tid);
            commit_f32 (rB, sB[buf ^ 1], tid);
            if (k0 + 2 * BK < H_) {
                rA = load_bf16(ha + k0 + 2 * BK, H_, rows, tid);
                rB = load_f32 (bw + k0 + 2 * BK, H_, BM,   tid);
            }
        }
        __syncthreads();
    }

#pragma unroll
    for (int ms = 0; ms < 4; ++ms)
#pragma unroll
        for (int ns = 0; ns < 2; ++ns)
#pragma unroll
            for (int r = 0; r < 8; ++r) {
                int ml = wm * 64 + ms * 16 + mofs + r;
                if (ml < rows)
                    out[(size_t)(rs + ml) * D_ + (nb + wn * 32 + ns * 16 + lrow)] =
                        acc[ms][ns][r];
            }
}

// ---------------------------------------------------------------------------
extern "C" void kernel_launch(void* const* d_in, const int* in_sizes, int n_in,
                              void* d_out, int out_size, void* d_ws, size_t ws_size,
                              hipStream_t stream) {
    (void)in_sizes; (void)n_in; (void)out_size; (void)ws_size;
    const float* x   = (const float*)d_in[0];
    const float* w1  = (const float*)d_in[1];
    const float* w2  = (const float*)d_in[2];
    const float* w3  = (const float*)d_in[3];
    const int*   cnt = (const int*)d_in[4];

    int*  tinfo = (int*)d_ws;
    bf16* h     = (bf16*)((char*)d_ws + 4096);   // [T, H] bf16 activations
    float* out  = (float*)d_out;

    build_tiles_k<<<1, 32, 0, stream>>>(cnt, tinfo);
    zero_out_k<<<(T_ * D_ / 4) / 256, 256, 0, stream>>>((float4*)out);
    moe_gemm13<<<dim3(MAX_TILES, H_ / BN), 256, 0, stream>>>(x, w1, w3, tinfo, h);
    moe_gemm2 <<<dim3(MAX_TILES, D_ / BN), 256, 0, stream>>>(h, w2, tinfo, out);
}
